// FasterRCNNModel_88450556494661
// MI455X (gfx1250) — matrix operand order repaired and checked by reading, not compiled, not run on hardware
//
#include <hip/hip_runtime.h>

typedef __attribute__((ext_vector_type(16))) __bf16 v16bf;
typedef __attribute__((ext_vector_type(8)))  float  v8f;

#define FH 64
#define FW 64
#define CDIM 512
#define NA 9
#define NPIX (FH*FW)          // 4096
#define NANCH (NPIX*NA)       // 36864
#define NPRE 6000
#define NPOST 300
#define MW 188                // ceil(6000/32)
#define POOLSZ 7
#define FLATD (CDIM*POOLSZ*POOLSZ)  // 25088
#define FVDIM 4096
#define NCLS 21
#define NREG 80
#define IOU_T 0.7f

// ---------------------------------------------------------------------------
// conv1: 3x3, 512->512, SAME, + bias + ReLU.  Implicit GEMM via bf16 WMMA.
// Each wave computes a 16(pixels) x 32(out-ch) tile: one A fragment feeds two
// WMMAs, halving the strided activation gathers per FLOP.
// Branchless zero-padding: clamp coordinates, multiply by 0/1 mask.
// Output y stored NHWC: y[pix*512 + oc].
// grid(2, 256) block(256)
// ---------------------------------------------------------------------------
__global__ __launch_bounds__(256) void conv1_wmma(
    const float* __restrict__ fm, const float* __restrict__ wgt,
    const float* __restrict__ bias, float* __restrict__ y)
{
  const int lane = threadIdx.x & 31;
  const int wave = threadIdx.x >> 5;
  const int n0 = blockIdx.x * 256 + wave * 32;   // first of two 16-ch tiles
  const int mt = blockIdx.y;                      // 0..255
  const int h  = mt >> 2;
  const int w0 = (mt & 3) * 16;
  const int l16 = lane & 15;
  const int hi  = lane >> 4;
  const int koffA = hi * 16;
  const int koffB = hi * 16;
  const int myw  = w0 + l16;            // A row -> pixel column
  const int myn0 = n0 + l16;            // B column, tile 0
  const int myn1 = myn0 + 16;           // B column, tile 1

  v8f acc0 = {};
  v8f acc1 = {};
  for (int rs = 0; rs < 9; ++rs) {
    const int r = rs / 3, s = rs % 3;
    const int hh = h + r - 1;
    const int ww = myw + s - 1;
    const int hc = min(max(hh, 0), FH - 1);       // clamped (always legal)
    const int wc = min(max(ww, 0), FW - 1);
    const float vmask =
        ((hh >= 0) && (hh < FH) && (ww >= 0) && (ww < FW)) ? 1.0f : 0.0f;
    const float* abase  = fm + hc * FW + wc;                      // + c*4096
    const float* bbase0 = wgt + (size_t)myn0 * (CDIM * 9) + rs;   // + c*9
    const float* bbase1 = wgt + (size_t)myn1 * (CDIM * 9) + rs;
    for (int c0 = 0; c0 < CDIM; c0 += 32) {
      v16bf va, vb0, vb1;
#pragma unroll
      for (int e = 0; e < 8; ++e) {
        const int c1 = c0 + koffA + e;
        const int c2 = c1 + 8;
        va[e]     = (__bf16)(abase[c1 * NPIX] * vmask);
        va[e + 8] = (__bf16)(abase[c2 * NPIX] * vmask);
      }
#pragma unroll
      for (int e = 0; e < 16; ++e) {
        vb0[e] = (__bf16)bbase0[(c0 + koffB + e) * 9];
        vb1[e] = (__bf16)bbase1[(c0 + koffB + e) * 9];
      }
      acc0 = __builtin_amdgcn_wmma_f32_16x16x32_bf16(
          false, va, false, vb0, (short)0, acc0, false, false);
      acc1 = __builtin_amdgcn_wmma_f32_16x16x32_bf16(
          false, va, false, vb1, (short)0, acc1, false, false);
    }
  }
  const float bv0 = bias[myn0];
  const float bv1 = bias[myn1];
#pragma unroll
  for (int rr = 0; rr < 8; ++rr) {
    const int m = rr + 8 * hi;                  // pixel within tile
    const int pix = h * FW + w0 + m;
    y[(size_t)pix * CDIM + myn0] = fmaxf(acc0[rr] + bv0, 0.0f);
    y[(size_t)pix * CDIM + myn1] = fmaxf(acc1[rr] + bv1, 0.0f);
  }
}

// ---------------------------------------------------------------------------
// RPN heads: 1x1 convs (9 cls + sigmoid, 36 box), y is NHWC.
// Activation row staged through LDS with the gfx1250 async-to-LDS path.
// grid(4096) block(256)
// ---------------------------------------------------------------------------
__global__ __launch_bounds__(256) void rpn_heads(
    const float* __restrict__ y,
    const float* __restrict__ cw, const float* __restrict__ cb,
    const float* __restrict__ bw, const float* __restrict__ bb,
    float* __restrict__ outScores, float* __restrict__ outDeltas)
{
  __shared__ float row[CDIM];
  const int pix = blockIdx.x;
  if (threadIdx.x < CDIM / 4) {
    // 128 lanes x 16B async copies: global -> LDS without touching VGPR data.
    unsigned ldsoff = (unsigned)(unsigned long long)(&row[threadIdx.x * 4]);
    unsigned long long gaddr =
        (unsigned long long)(y + (size_t)pix * CDIM + threadIdx.x * 4);
    asm volatile("global_load_async_to_lds_b128 %0, %1, off"
                 :: "v"(ldsoff), "v"(gaddr) : "memory");
  }
  asm volatile("s_wait_asynccnt 0x0" ::: "memory");
  __syncthreads();
  for (int o = threadIdx.x; o < NA + NA * 4; o += blockDim.x) {
    if (o < NA) {
      const float* w = cw + (size_t)o * CDIM;
      float d = cb[o];
      for (int c = 0; c < CDIM; ++c) d += row[c] * w[c];
      outScores[pix * NA + o] = 1.0f / (1.0f + __expf(-d));
    } else {
      const int dd = o - NA;
      const float* w = bw + (size_t)dd * CDIM;
      float d = bb[dd];
      for (int c = 0; c < CDIM; ++c) d += row[c] * w[c];
      outDeltas[pix * (NA * 4) + dd] = d;
    }
  }
}

// ---------------------------------------------------------------------------
// Anchor decode + clip -> boxes (y1,x1,y2,x2)
// ---------------------------------------------------------------------------
__global__ void decode_boxes(const float* __restrict__ amap,
                             const float* __restrict__ deltas,
                             float* __restrict__ boxes)
{
  const int i = blockIdx.x * 256 + threadIdx.x;
  if (i >= NANCH) return;
  const int pix = i / NA, a = i % NA;
  const float* an = amap + pix * (NA * 4) + a * 4;
  const float* dl = deltas + pix * (NA * 4) + a * 4;
  const float cy = an[0] + dl[0] * an[2];
  const float cx = an[1] + dl[1] * an[3];
  const float sh = an[2] * expf(dl[2]);
  const float sw = an[3] * expf(dl[3]);
  boxes[i * 4 + 0] = fminf(fmaxf(cy - 0.5f * sh, 0.0f), 1023.0f);
  boxes[i * 4 + 1] = fminf(fmaxf(cx - 0.5f * sw, 0.0f), 1023.0f);
  boxes[i * 4 + 2] = fminf(fmaxf(cy + 0.5f * sh, 0.0f), 1023.0f);
  boxes[i * 4 + 3] = fminf(fmaxf(cx + 0.5f * sw, 0.0f), 1023.0f);
}

// ---------------------------------------------------------------------------
// Exact top-6000 by rank counting (stable, matches lax.top_k ordering).
// grid(144) block(256)
// ---------------------------------------------------------------------------
__global__ __launch_bounds__(256) void topk_pre(
    const float* __restrict__ scores, const float* __restrict__ boxes,
    float* __restrict__ sc_pre, float* __restrict__ bx)
{
  __shared__ float tile[256];
  const int i = blockIdx.x * 256 + threadIdx.x;
  const float si = scores[i];
  int rank = 0;
  for (int t0 = 0; t0 < NANCH; t0 += 256) {
    __syncthreads();
    tile[threadIdx.x] = scores[t0 + threadIdx.x];
    __syncthreads();
#pragma unroll 8
    for (int j = 0; j < 256; ++j) {
      const float sj = tile[j];
      const int jj = t0 + j;
      rank += (sj > si) || (sj == si && jj < i);
    }
  }
  if (rank < NPRE) {
    sc_pre[rank] = si;
    bx[rank * 4 + 0] = boxes[i * 4 + 0];
    bx[rank * 4 + 1] = boxes[i * 4 + 1];
    bx[rank * 4 + 2] = boxes[i * 4 + 2];
    bx[rank * 4 + 3] = boxes[i * 4 + 3];
  }
}

// ---------------------------------------------------------------------------
// NMS suppression bitmask: mask[i][wj] bit t = (j=32*wj+t > i) && iou(i,j)>T
// grid(188, 24) block(256)
// ---------------------------------------------------------------------------
__global__ __launch_bounds__(256) void nms_mask(
    const float* __restrict__ bx, unsigned* __restrict__ mask)
{
  const int i = blockIdx.y * blockDim.x + threadIdx.x;
  const int wj = blockIdx.x;
  if (i >= NPRE) return;
  const float a0 = bx[i * 4 + 0], a1 = bx[i * 4 + 1];
  const float a2 = bx[i * 4 + 2], a3 = bx[i * 4 + 3];
  const float areaI = (a2 - a0) * (a3 - a1);
  unsigned bits = 0;
  const int jbase = wj * 32;
#pragma unroll 4
  for (int t = 0; t < 32; ++t) {
    const int j = jbase + t;
    if (j >= NPRE || j <= i) continue;
    const float b0 = bx[j * 4 + 0], b1 = bx[j * 4 + 1];
    const float b2 = bx[j * 4 + 2], b3 = bx[j * 4 + 3];
    const float areaJ = (b2 - b0) * (b3 - b1);
    const float yy1 = fmaxf(a0, b0), xx1 = fmaxf(a1, b1);
    const float yy2 = fminf(a2, b2), xx2 = fminf(a3, b3);
    const float inter = fmaxf(yy2 - yy1, 0.0f) * fmaxf(xx2 - xx1, 0.0f);
    const float iou = inter / (areaI + areaJ - inter + 1e-8f);
    if (iou > IOU_T) bits |= (1u << t);
  }
  mask[(size_t)i * MW + wj] = bits;
}

// Serial scan, single wave32. Reproduces the reference fori_loop exactly.
__global__ void nms_scan(const unsigned* __restrict__ mask, int* __restrict__ sup)
{
  __shared__ unsigned remv[MW];
  for (int w = threadIdx.x; w < MW; w += 32) remv[w] = 0;
  __syncthreads();
  for (int i = 0; i < NPRE; ++i) {
    const unsigned wv = remv[i >> 5];
    const int bit = (wv >> (i & 31)) & 1;
    if (threadIdx.x == 0) sup[i] = bit;
    if (!bit) {
      for (int w = threadIdx.x; w < MW; w += 32)
        remv[w] |= mask[(size_t)i * MW + w];
    }
  }
}

// ---------------------------------------------------------------------------
// Top-300 of kept scores; emit proposals (y1,x1,y2,x2) and rois (x1,y1,x2,y2).
// grid(24) block(256)
// ---------------------------------------------------------------------------
__global__ __launch_bounds__(256) void topk_post(
    const float* __restrict__ sc_pre, const int* __restrict__ sup,
    const float* __restrict__ bx,
    float* __restrict__ proposals, float* __restrict__ rois)
{
  __shared__ float tile[256];
  const int i = blockIdx.x * 256 + threadIdx.x;
  const float si = (i < NPRE && !sup[i]) ? sc_pre[i] : -INFINITY;
  int rank = 0;
  for (int t0 = 0; t0 < NPRE; t0 += 256) {
    __syncthreads();
    const int idx = t0 + threadIdx.x;
    tile[threadIdx.x] = (idx < NPRE && !sup[idx]) ? sc_pre[idx] : -INFINITY;
    __syncthreads();
    const int jmax = (NPRE - t0 < 256) ? (NPRE - t0) : 256;
    for (int j = 0; j < jmax; ++j) {
      const float sj = tile[j];
      const int jj = t0 + j;
      rank += (sj > si) || (sj == si && jj < i);
    }
  }
  if (i < NPRE && rank < NPOST) {
    const float y1 = bx[i * 4 + 0], x1 = bx[i * 4 + 1];
    const float y2 = bx[i * 4 + 2], x2 = bx[i * 4 + 3];
    proposals[rank * 4 + 0] = y1; proposals[rank * 4 + 1] = x1;
    proposals[rank * 4 + 2] = y2; proposals[rank * 4 + 3] = x2;
    rois[rank * 4 + 0] = x1; rois[rank * 4 + 1] = y1;
    rois[rank * 4 + 2] = x2; rois[rank * 4 + 3] = y2;
  }
}

// ---------------------------------------------------------------------------
// ROI max-pool 7x7 over 512 channels. grid(300) block(256)
// pooled[roi][c*49 + ph*7 + pw]
// ---------------------------------------------------------------------------
__global__ __launch_bounds__(256) void roi_pool(
    const float* __restrict__ fm, const float* __restrict__ rois,
    float* __restrict__ pooled)
{
  const int roi = blockIdx.x;
  const float s = 1.0f / 16.0f;
  const float x1 = rintf(rois[roi * 4 + 0] * s);
  const float y1 = rintf(rois[roi * 4 + 1] * s);
  const float x2 = rintf(rois[roi * 4 + 2] * s);
  const float y2 = rintf(rois[roi * 4 + 3] * s);
  const float rw = fmaxf(x2 - x1 + 1.0f, 1.0f);
  const float rh = fmaxf(y2 - y1 + 1.0f, 1.0f);
  const float bh = rh / (float)POOLSZ;
  const float bw = rw / (float)POOLSZ;
  for (int cell = threadIdx.x; cell < CDIM * POOLSZ * POOLSZ; cell += blockDim.x) {
    const int c = cell / (POOLSZ * POOLSZ);
    const int pq = cell % (POOLSZ * POOLSZ);
    const int ph = pq / POOLSZ, pw = pq % POOLSZ;
    const int hs = (int)fminf(fmaxf(floorf(ph * bh) + y1, 0.0f), (float)FH);
    const int he = (int)fminf(fmaxf(ceilf((ph + 1) * bh) + y1, 0.0f), (float)FH);
    const int wsb = (int)fminf(fmaxf(floorf(pw * bw) + x1, 0.0f), (float)FW);
    const int we = (int)fminf(fmaxf(ceilf((pw + 1) * bw) + x1, 0.0f), (float)FW);
    float m = -INFINITY;
    for (int hy = hs; hy < he; ++hy)
      for (int wx = wsb; wx < we; ++wx)
        m = fmaxf(m, fm[(size_t)c * NPIX + hy * FW + wx]);
    pooled[(size_t)roi * FLATD + cell] = (hs < he && wsb < we) ? m : 0.0f;
  }
}

// ---------------------------------------------------------------------------
// Generic WMMA GEMM: Out[M,N] = A[M,K] * W[N,K]^T + bias (+ReLU).
// Each wave computes a 16x32 tile (2 accumulators, A fragment reused).
// grid(ceil(N/256), ceil(M/16)) block(256).
// ---------------------------------------------------------------------------
__global__ __launch_bounds__(256) void gemm_wmma(
    const float* __restrict__ A, const float* __restrict__ W,
    const float* __restrict__ bias, float* __restrict__ Out,
    int M, int N, int K, int relu)
{
  const int lane = threadIdx.x & 31;
  const int wave = threadIdx.x >> 5;
  const int n0 = blockIdx.x * 256 + wave * 32;   // first of two 16-col tiles
  const int m0 = blockIdx.y * 16;
  const int l16 = lane & 15;
  const int hi  = lane >> 4;
  int arow = m0 + l16; if (arow >= M) arow = M - 1;
  int brow0 = n0 + l16;      if (brow0 >= N) brow0 = N - 1;
  int brow1 = n0 + 16 + l16; if (brow1 >= N) brow1 = N - 1;
  const float* Ab  = A + (size_t)arow * K + hi * 16;
  const float* Wb0 = W + (size_t)brow0 * K + hi * 16;
  const float* Wb1 = W + (size_t)brow1 * K + hi * 16;
  v8f acc0 = {};
  v8f acc1 = {};
  for (int k0 = 0; k0 < K; k0 += 32) {
    const float* ap  = Ab + k0;
    const float* bp0 = Wb0 + k0;
    const float* bp1 = Wb1 + k0;
    __builtin_prefetch(bp0 + 256, 0, 1);
    __builtin_prefetch(bp1 + 256, 0, 1);
    v16bf va, vb0, vb1;
#pragma unroll
    for (int e = 0; e < 8; ++e) {
      va[e]     = (__bf16)ap[e];
      va[e + 8] = (__bf16)ap[e + 8];
    }
#pragma unroll
    for (int e = 0; e < 16; ++e) {
      vb0[e] = (__bf16)bp0[e];
      vb1[e] = (__bf16)bp1[e];
    }
    acc0 = __builtin_amdgcn_wmma_f32_16x16x32_bf16(
        false, va, false, vb0, (short)0, acc0, false, false);
    acc1 = __builtin_amdgcn_wmma_f32_16x16x32_bf16(
        false, va, false, vb1, (short)0, acc1, false, false);
  }
  const int na = n0 + l16;
  const int nb = n0 + 16 + l16;
  const float bva = (na < N) ? bias[na] : 0.0f;
  const float bvb = (nb < N) ? bias[nb] : 0.0f;
#pragma unroll
  for (int r = 0; r < 8; ++r) {
    const int m = m0 + r + 8 * hi;
    if (m < M) {
      if (na < N) {
        float v = acc0[r] + bva;
        if (relu) v = fmaxf(v, 0.0f);
        Out[(size_t)m * N + na] = v;
      }
      if (nb < N) {
        float v = acc1[r] + bvb;
        if (relu) v = fmaxf(v, 0.0f);
        Out[(size_t)m * N + nb] = v;
      }
    }
  }
}

// 21-way softmax per row. grid(300) block(32)
__global__ void softmax21(const float* __restrict__ logits, float* __restrict__ out)
{
  const int m = blockIdx.x;
  const int lane = threadIdx.x;
  float v = (lane < NCLS) ? logits[m * NCLS + lane] : -INFINITY;
  float mx = v;
  for (int o = 16; o > 0; o >>= 1) mx = fmaxf(mx, __shfl_xor(mx, o, 32));
  float e = (lane < NCLS) ? expf(v - mx) : 0.0f;
  float ssum = e;
  for (int o = 16; o > 0; o >>= 1) ssum += __shfl_xor(ssum, o, 32);
  if (lane < NCLS) out[m * NCLS + lane] = e / ssum;
}

// ---------------------------------------------------------------------------
extern "C" void kernel_launch(void* const* d_in, const int* in_sizes, int n_in,
                              void* d_out, int out_size, void* d_ws, size_t ws_size,
                              hipStream_t stream)
{
  const float* fm   = (const float*)d_in[0];
  const float* amap = (const float*)d_in[1];
  const float* c1w  = (const float*)d_in[3];
  const float* c1b  = (const float*)d_in[4];
  const float* clw  = (const float*)d_in[5];
  const float* clb  = (const float*)d_in[6];
  const float* bxw  = (const float*)d_in[7];
  const float* bxb  = (const float*)d_in[8];
  const float* f1w  = (const float*)d_in[9];
  const float* f1b  = (const float*)d_in[10];
  const float* f2w  = (const float*)d_in[11];
  const float* f2b  = (const float*)d_in[12];
  const float* cwv  = (const float*)d_in[13];
  const float* cbv  = (const float*)d_in[14];
  const float* rwv  = (const float*)d_in[15];
  const float* rbv  = (const float*)d_in[16];

  float* out = (float*)d_out;
  float* outScores = out;                         // 36864
  float* outDeltas = out + NANCH;                 // 147456
  float* outProps  = out + NANCH + NANCH * 4;     // offset 184320, 1200
  float* outCls    = outProps + NPOST * 4;        // offset 185520, 6300
  float* outReg    = outCls + NPOST * NCLS;       // offset 191820, 24000

  char* wsb = (char*)d_ws;
  size_t off = 0;
  auto wsalloc = [&](size_t bytes) {
    size_t o = off;
    off = (off + bytes + 255) & ~(size_t)255;
    return (void*)(wsb + o);
  };
  float*    y       = (float*)   wsalloc((size_t)NPIX * CDIM * 4);
  float*    boxes   = (float*)   wsalloc((size_t)NANCH * 4 * 4);
  float*    sc_pre  = (float*)   wsalloc((size_t)NPRE * 4);
  float*    bx      = (float*)   wsalloc((size_t)NPRE * 4 * 4);
  unsigned* mask    = (unsigned*)wsalloc((size_t)NPRE * MW * 4);
  int*      sup     = (int*)     wsalloc((size_t)NPRE * 4);
  float*    rois    = (float*)   wsalloc((size_t)NPOST * 4 * 4);
  float*    pooled  = (float*)   wsalloc((size_t)NPOST * FLATD * 4);
  float*    h1      = (float*)   wsalloc((size_t)NPOST * FVDIM * 4);
  float*    h2      = (float*)   wsalloc((size_t)NPOST * FVDIM * 4);
  float*    logits  = (float*)   wsalloc((size_t)NPOST * NCLS * 4);

  // 1) conv1 + relu (WMMA implicit GEMM, 16x32 per wave)
  conv1_wmma<<<dim3(2, 256), dim3(256), 0, stream>>>(fm, c1w, c1b, y);
  // 2) rpn heads (sigmoid cls + box deltas) -> d_out regions
  rpn_heads<<<dim3(NPIX), dim3(256), 0, stream>>>(y, clw, clb, bxw, bxb,
                                                  outScores, outDeltas);
  // 3) decode anchors -> clipped boxes
  decode_boxes<<<dim3(NANCH / 256), dim3(256), 0, stream>>>(amap, outDeltas, boxes);
  // 4) exact top-6000
  topk_pre<<<dim3(NANCH / 256), dim3(256), 0, stream>>>(outScores, boxes, sc_pre, bx);
  // 5) NMS bitmask + serial scan
  nms_mask<<<dim3(MW, (NPRE + 255) / 256), dim3(256), 0, stream>>>(bx, mask);
  nms_scan<<<dim3(1), dim3(32), 0, stream>>>(mask, sup);
  // 6) top-300 -> proposals + rois
  topk_post<<<dim3((NPRE + 255) / 256), dim3(256), 0, stream>>>(
      sc_pre, sup, bx, outProps, rois);
  // 7) ROI max-pool on the raw feature map
  roi_pool<<<dim3(NPOST), dim3(256), 0, stream>>>(fm, rois, pooled);
  // 8) FC stack (WMMA GEMMs, 16x32 per wave)
  gemm_wmma<<<dim3(FVDIM / 256, (NPOST + 15) / 16), dim3(256), 0, stream>>>(
      pooled, f1w, f1b, h1, NPOST, FVDIM, FLATD, 1);
  gemm_wmma<<<dim3(FVDIM / 256, (NPOST + 15) / 16), dim3(256), 0, stream>>>(
      h1, f2w, f2b, h2, NPOST, FVDIM, FVDIM, 1);
  gemm_wmma<<<dim3(1, (NPOST + 15) / 16), dim3(256), 0, stream>>>(
      h2, cwv, cbv, logits, NPOST, NCLS, FVDIM, 0);
  gemm_wmma<<<dim3(1, (NPOST + 15) / 16), dim3(256), 0, stream>>>(
      h2, rwv, rbv, outReg, NPOST, NREG, FVDIM, 0);
  // 9) softmax over classes
  softmax21<<<dim3(NPOST), dim3(32), 0, stream>>>(logits, outCls);

  (void)in_sizes; (void)n_in; (void)out_size; (void)ws_size;
}
